// RuleNet_19670950216266
// MI455X (gfx1250) — matrix operand order repaired and checked by reading, not compiled, not run on hardware
//
#include <hip/hip_runtime.h>
#include <hip/hip_bf16.h>

// ---------------- problem constants ----------------
#define B_      1024
#define T_      120
#define SENS_   65
#define H_      512
#define RULE_   64
#define NTASK_  50
#define ENCH_   128
#define OUT_    33
#define KIN_    160          // RULE+SENSORY = 129, padded to 5*32
#define KH_     512
#define KTOT_   672          // 512 + 160
#define NGATE_  1536

typedef __attribute__((ext_vector_type(16))) __bf16 v16bf;
typedef __attribute__((ext_vector_type(8)))  float  v8f;
typedef __attribute__((ext_vector_type(4)))  unsigned int u32x4;
typedef __attribute__((ext_vector_type(8)))  int i32x8;
typedef __attribute__((ext_vector_type(4)))  int i32x4;

#define WMMA_BF16(A, Bm, C) \
  __builtin_amdgcn_wmma_f32_16x16x32_bf16(false, (A), false, (Bm), (short)0, (C), false, false)

// Tensor Data Mover availability (device pass only; arity differs by toolchain)
#if defined(__HIP_DEVICE_COMPILE__) && __has_builtin(__builtin_amdgcn_tensor_load_to_lds) && \
    __has_builtin(__builtin_amdgcn_s_wait_tensorcnt)
#define TDM_OK 1
#else
#define TDM_OK 0
#endif

__device__ __forceinline__ float sigmoidf_(float x) {
  return 1.0f / (1.0f + __expf(-x));
}

#if TDM_OK
// Issue one TDM 2-D tile load: tile1 rows x tile0 cols of 2-byte elements,
// row stride stride0 (elements), packed row-major into LDS at lds_addr.
// D# per CDNA5 ISA 8.3/8.4: group0 {count=1, lds_addr, global_addr, type=2},
// group1 {data_size=1(2B), tensor_dim0/1, tile_dim0/1, tensor_dim0_stride}.
__device__ __forceinline__ void tdm_load_2d_bf16(const void* gsrc, unsigned lds_addr,
                                                 unsigned td0, unsigned td1,
                                                 unsigned tile0, unsigned tile1,
                                                 unsigned stride0) {
  const unsigned long long ga = (unsigned long long)(uintptr_t)gsrc;
  u32x4 g0;
  g0[0] = 1u;                                                   // count=1, user D#
  g0[1] = lds_addr;                                             // LDS byte address
  g0[2] = (unsigned)ga;                                         // global_addr[31:0]
  g0[3] = (unsigned)((ga >> 32) & 0x01FFFFFFu) | 0x80000000u;   // global_addr[56:32] | type=2
  i32x8 g1;
  g1[0] = 0x00010000;                                           // data_size=1 (2 bytes)
  g1[1] = (int)((td0 & 0xFFFFu) << 16);                         // tensor_dim0[15:0]
  g1[2] = (int)((td0 >> 16) | ((td1 & 0xFFFFu) << 16));         // tensor_dim0[31:16] | tensor_dim1[15:0]
  g1[3] = (int)((td1 >> 16) | (tile0 << 16));                   // tensor_dim1[31:16] | tile_dim0
  g1[4] = (int)(tile1 & 0xFFFFu);                               // tile_dim1 (tile_dim2=0)
  g1[5] = (int)stride0;                                         // tensor_dim0_stride[31:0]
  g1[6] = 0;                                                    // stride0[47:32] | stride1[15:0]
  g1[7] = 0;
  i32x4 gz = {0, 0, 0, 0};
#if __clang_major__ >= 23
  i32x8 gz8 = {0, 0, 0, 0, 0, 0, 0, 0};
  __builtin_amdgcn_tensor_load_to_lds(g0, g1, gz, gz, gz8, 0);
#else
  __builtin_amdgcn_tensor_load_to_lds(g0, g1, gz, gz, 0);
#endif
}
#endif

// A fragment (16x32 bf16, M x K), source row-major [16][stride] in LDS.
__device__ __forceinline__ v16bf load_frag_a(const __bf16* base, int stride, int lane) {
  int m  = lane & 15;
  int k0 = (lane >> 4) << 3;   // 0 or 8
  union { uint4 u[2]; v16bf v; } r;
  r.u[0] = *(const uint4*)(base + m * stride + k0);
  r.u[1] = *(const uint4*)(base + m * stride + k0 + 16);
  return r.v;
}

// B fragment (32x16 bf16, K x N), source stored as W[n][k] row-major [16][stride].
__device__ __forceinline__ v16bf load_frag_b(const __bf16* base, int stride, int lane) {
  int n  = lane & 15;
  int k0 = (lane >> 4) << 3;   // 0 or 8
  union { uint4 u[2]; v16bf v; } r;
  const __bf16* p = base + n * stride + k0;
  r.u[0] = *(const uint4*)(p);
  r.u[1] = *(const uint4*)(p + 16);
  return r.v;
}

// ---------------- rule encoder MLP (tiny) ----------------
__global__ __launch_bounds__(256) void rule_enc_kernel(
    const float* __restrict__ task_rule, const float* __restrict__ rule_tf,
    const float* __restrict__ ew1, const float* __restrict__ eb1,
    const float* __restrict__ ew2, const float* __restrict__ eb2,
    float* __restrict__ renc) {
  __shared__ float ruleT[8][RULE_];
  __shared__ float h1s[8][ENCH_];
  const int lane = threadIdx.x & 31, wave = threadIdx.x >> 5;
  const int b = blockIdx.x * 8 + wave;
  for (int f = lane; f < RULE_; f += 32) {
    float s = 0.f;
    for (int nt = 0; nt < NTASK_; ++nt)
      s += task_rule[b * NTASK_ + nt] * rule_tf[nt * RULE_ + f];
    ruleT[wave][f] = s;
  }
  __syncthreads();
  for (int f = lane; f < ENCH_; f += 32) {
    float s = eb1[f];
    for (int k = 0; k < RULE_; ++k) s += ruleT[wave][k] * ew1[f * RULE_ + k];
    h1s[wave][f] = fmaxf(s, 0.f);
  }
  __syncthreads();
  for (int f = lane; f < RULE_; f += 32) {
    float s = eb2[f];
    for (int k = 0; k < ENCH_; ++k) s += h1s[wave][k] * ew2[f * ENCH_ + k];
    renc[b * RULE_ + f] = fmaxf(s, 0.f);
  }
}

// ---------------- weight prep: [w_hh | w_ih pad] -> bf16, out_w -> bf16 padded ----------------
__global__ void prep_weights(const float* __restrict__ w_ih, const float* __restrict__ w_hh,
                             const float* __restrict__ out_w,
                             __bf16* __restrict__ Wc, __bf16* __restrict__ OutW) {
  const int stride = gridDim.x * blockDim.x;
  const int idx = blockIdx.x * blockDim.x + threadIdx.x;
  const int total1 = NGATE_ * KTOT_;
  for (int i = idx; i < total1; i += stride) {
    int g = i / KTOT_, k = i % KTOT_;
    float v;
    if (k < KH_)              v = w_hh[g * KH_ + k];
    else if ((k - KH_) < 129) v = w_ih[g * 129 + (k - KH_)];
    else                      v = 0.f;
    Wc[i] = (__bf16)v;
  }
  const int total2 = 48 * H_;
  for (int i = idx; i < total2; i += stride) {
    int o = i / H_, k = i % H_;
    OutW[i] = (__bf16)((o < OUT_) ? out_w[o * H_ + k] : 0.f);
  }
}

// ---------------- build padded bf16 rnn_ins: [rule_enc*mask(64) | x(65) | 0 pad] ----------------
__global__ void prep_x(const float* __restrict__ x, const float* __restrict__ renc,
                       __bf16* __restrict__ Xbf) {
  const long total = (long)B_ * T_ * KIN_;
  const long stride = (long)gridDim.x * blockDim.x;
  for (long i = (long)blockIdx.x * blockDim.x + threadIdx.x; i < total; i += stride) {
    int k = (int)(i % KIN_);
    long bt = i / KIN_;
    int t = (int)(bt % T_);
    long b = bt / T_;
    float v;
    if (k < RULE_)    v = (t < 20) ? renc[b * RULE_ + k] : 0.f;   // INFO_ONSET=0, INFO_DUR=20
    else if (k < 129) v = x[bt * SENS_ + (k - RULE_)];
    else              v = 0.f;
    Xbf[i] = (__bf16)v;
  }
}

// ---------------- h0 = 0.1 (bf16 ping buffer) ----------------
__global__ void init_h(__bf16* __restrict__ Hb) {
  const long total = (long)B_ * H_;
  const long stride = (long)gridDim.x * blockDim.x;
  for (long i = (long)blockIdx.x * blockDim.x + threadIdx.x; i < total; i += stride)
    Hb[i] = (__bf16)0.1f;
}

// ---------------- one GRU timestep ----------------
// gates = [h_{t-1} | rnn_ins_t] @ [w_hh | w_ih]^T via bf16 WMMA, f32 acc.
// Double-buffered LDS; weight tiles staged by the Tensor Data Mover (wave 0
// issues 3 TDM 64x32 tile loads per k-tile), A tiles by vector loads.
__global__ __launch_bounds__(256) void gru_step(
    const __bf16* __restrict__ Wc,      // [1536][672]
    const __bf16* __restrict__ Xbf,     // [B][T][160]
    const __bf16* __restrict__ Hb_prev, // [B][512]
    __bf16* __restrict__ Hb_next,       // [B][512]
    const float* __restrict__ b_ih, const float* __restrict__ b_hh,
    float* __restrict__ hid,            // [B][T][512] (d_out region)
    int t) {
  __shared__ __align__(16) __bf16 Alds[2][64 * 32];
  __shared__ __align__(16) __bf16 Wlds[2][192 * 32];

  const int tid = threadIdx.x;
  const int lane = tid & 31, wave = tid >> 5;
  const int waveM = wave & 3, waveN = wave >> 2;
  const int m0 = blockIdx.x * 64;   // batch tile
  const int n0 = blockIdx.y * 64;   // hidden-col tile

  v8f accR[2] = {}, accZ[2] = {}, accHN[2] = {}, accIN[2] = {};

  const int arow = tid >> 2;        // 0..63
  const int akk  = (tid & 3) << 3;  // 0,8,16,24

  auto stage_a = [&](int kt, int buf) {
    const int kg = kt * 32 + akk;
    const __bf16* src;
    if (kg < KH_) src = Hb_prev + (size_t)(m0 + arow) * KH_ + kg;
    else          src = Xbf + ((size_t)(m0 + arow) * T_ + t) * KIN_ + (kg - KH_);
    *(uint4*)(&Alds[buf][arow * 32 + akk]) = *(const uint4*)src;
  };
#if TDM_OK
  auto issue_w_tdm = [&](int kt, int buf) {
    for (int g = 0; g < 3; ++g) {   // r / z / n gate blocks: 64 rows x 32 cols
      const __bf16* gsrc = Wc + (size_t)(g * KH_ + n0) * KTOT_ + kt * 32;
      const unsigned lds_addr = (unsigned)(uintptr_t)(&Wlds[buf][g * 64 * 32]);
      tdm_load_2d_bf16(gsrc, lds_addr, KTOT_, 64, 32, 64, KTOT_);
    }
  };
#else
  auto stage_w = [&](int kt, int buf) {
    for (int it = 0; it < 3; ++it) {
      const int f = it * 2048 + tid * 8;
      const int r = f >> 5, kk = f & 31;
      const int gate = r >> 6, nloc = r & 63;
      const __bf16* src = Wc + (size_t)(gate * KH_ + n0 + nloc) * KTOT_ + kt * 32 + kk;
      *(uint4*)(&Wlds[buf][r * 32 + kk]) = *(const uint4*)src;
    }
  };
#endif

  // prologue: tile 0 into buffer 0
  stage_a(0, 0);
#if TDM_OK
  if (wave == 0) issue_w_tdm(0, 0);
#else
  stage_w(0, 0);
#endif

  for (int kt = 0; kt < 21; ++kt) {
    const int cur = kt & 1, nxt = cur ^ 1;
#if TDM_OK
    if (wave == 0) __builtin_amdgcn_s_wait_tensorcnt(0);   // tile kt landed in LDS
#endif
    __syncthreads();
    if (kt < 20) {                                         // overlap stage kt+1 with compute kt
#if TDM_OK
      if (wave == 0) issue_w_tdm(kt + 1, nxt);
#else
      stage_w(kt + 1, nxt);
#endif
      stage_a(kt + 1, nxt);
    }

    const v16bf a   = load_frag_a(&Alds[cur][waveM * 16 * 32], 32, lane);
    const int ns0 = waveN * 32, ns1 = waveN * 32 + 16;
    const v16bf bR0 = load_frag_b(&Wlds[cur][(0 * 64 + ns0) * 32], 32, lane);
    const v16bf bZ0 = load_frag_b(&Wlds[cur][(1 * 64 + ns0) * 32], 32, lane);
    const v16bf bN0 = load_frag_b(&Wlds[cur][(2 * 64 + ns0) * 32], 32, lane);
    const v16bf bR1 = load_frag_b(&Wlds[cur][(0 * 64 + ns1) * 32], 32, lane);
    const v16bf bZ1 = load_frag_b(&Wlds[cur][(1 * 64 + ns1) * 32], 32, lane);
    const v16bf bN1 = load_frag_b(&Wlds[cur][(2 * 64 + ns1) * 32], 32, lane);

    accR[0] = WMMA_BF16(a, bR0, accR[0]);
    accZ[0] = WMMA_BF16(a, bZ0, accZ[0]);
    accR[1] = WMMA_BF16(a, bR1, accR[1]);
    accZ[1] = WMMA_BF16(a, bZ1, accZ[1]);
    if (kt < 16) {                       // hidden part of candidate gate
      accHN[0] = WMMA_BF16(a, bN0, accHN[0]);
      accHN[1] = WMMA_BF16(a, bN1, accHN[1]);
    } else {                             // input part of candidate gate
      accIN[0] = WMMA_BF16(a, bN0, accIN[0]);
      accIN[1] = WMMA_BF16(a, bN1, accIN[1]);
    }
    __syncthreads();
  }

  // fused GRU gating epilogue
#pragma unroll
  for (int s = 0; s < 2; ++s) {
    const int n = n0 + waveN * 32 + s * 16 + (lane & 15);
    const float bihR = b_ih[n],          bhhR = b_hh[n];
    const float bihZ = b_ih[H_ + n],     bhhZ = b_hh[H_ + n];
    const float bihN = b_ih[2 * H_ + n], bhhN = b_hh[2 * H_ + n];
    const int mrow = m0 + waveM * 16 + ((lane >> 4) << 3);
#pragma unroll
    for (int j = 0; j < 8; ++j) {
      const int m = mrow + j;
      const float r  = sigmoidf_(accR[s][j] + bihR + bhhR);
      const float z  = sigmoidf_(accZ[s][j] + bihZ + bhhZ);
      const float nn = fmaxf(accIN[s][j] + bihN + r * (accHN[s][j] + bhhN), 0.f);
      const float hp = (t == 0) ? 0.1f
                                : hid[((size_t)m * T_ + (t - 1)) * H_ + n];
      const float hnew = (1.f - z) * nn + z * hp;
      hid[((size_t)m * T_ + t) * H_ + n] = hnew;
      Hb_next[(size_t)m * H_ + n] = (__bf16)hnew;
    }
  }
}

// ---------------- output projection: sigmoid(hid @ out_w^T + b), N padded 33->48 ----------------
__global__ __launch_bounds__(256) void out_proj(
    const float* __restrict__ hid, const __bf16* __restrict__ OutW,
    const float* __restrict__ out_b, float* __restrict__ out) {
  __shared__ __align__(16) __bf16 Wlds[48 * H_];   // 48 KB, staged once (TDM)
  __shared__ __align__(16) __bf16 Alds[128 * 32];  // 8 KB per k-tile

  const int tid = threadIdx.x, lane = tid & 31, wave = tid >> 5;
  const size_t bt0 = (size_t)blockIdx.x * 128;

#if TDM_OK
  if (wave == 0) {
    tdm_load_2d_bf16(OutW, (unsigned)(uintptr_t)(&Wlds[0]), H_, 48, H_, 48, H_);
    __builtin_amdgcn_s_wait_tensorcnt(0);
  }
#else
  for (int it = 0; it < 12; ++it) {                // 48*512 bf16
    const int f = it * 2048 + tid * 8;
    *(uint4*)(Wlds + f) = *(const uint4*)(OutW + f);
  }
#endif

  v8f acc[3] = {};
  for (int kt = 0; kt < 16; ++kt) {
    for (int g2 = 0; g2 < 2; ++g2) {               // stage A: 128x32 f32 -> bf16
      const int f = g2 * 2048 + tid * 8;
      const int row = f >> 5, kk = f & 31;
      const float* src = hid + (bt0 + row) * H_ + kt * 32 + kk;
      const float4 x0 = *(const float4*)src;
      const float4 x1 = *(const float4*)(src + 4);
      __bf16* dst = Alds + row * 32 + kk;
      dst[0] = (__bf16)x0.x; dst[1] = (__bf16)x0.y;
      dst[2] = (__bf16)x0.z; dst[3] = (__bf16)x0.w;
      dst[4] = (__bf16)x1.x; dst[5] = (__bf16)x1.y;
      dst[6] = (__bf16)x1.z; dst[7] = (__bf16)x1.w;
    }
    __syncthreads();
    const v16bf a = load_frag_a(Alds + wave * 16 * 32, 32, lane);
#pragma unroll
    for (int s = 0; s < 3; ++s) {
      const v16bf b = load_frag_b(Wlds + s * 16 * H_ + kt * 32, H_, lane);
      acc[s] = WMMA_BF16(a, b, acc[s]);
    }
    __syncthreads();
  }

#pragma unroll
  for (int s = 0; s < 3; ++s) {
    const int o = s * 16 + (lane & 15);
    const bool valid = (o < OUT_);
    const float bias = valid ? out_b[o] : 0.f;
    const int mbase = wave * 16 + ((lane >> 4) << 3);
#pragma unroll
    for (int j = 0; j < 8; ++j) {
      const size_t bt = bt0 + mbase + j;
      if (valid) out[bt * OUT_ + o] = sigmoidf_(acc[s][j] + bias);
    }
  }
}

// ---------------- launch ----------------
extern "C" void kernel_launch(void* const* d_in, const int* in_sizes, int n_in,
                              void* d_out, int out_size, void* d_ws, size_t ws_size,
                              hipStream_t stream) {
  const float* x         = (const float*)d_in[0];
  const float* task_rule = (const float*)d_in[1];
  const float* rule_tf   = (const float*)d_in[2];
  const float* ew1       = (const float*)d_in[3];
  const float* eb1       = (const float*)d_in[4];
  const float* ew2       = (const float*)d_in[5];
  const float* eb2       = (const float*)d_in[6];
  const float* w_ih      = (const float*)d_in[7];
  const float* w_hh      = (const float*)d_in[8];
  const float* b_ih      = (const float*)d_in[9];
  const float* b_hh      = (const float*)d_in[10];
  const float* out_w     = (const float*)d_in[11];
  const float* out_b     = (const float*)d_in[12];

  float* out  = (float*)d_out;                       // [B][T][33]
  float* hid  = out + (size_t)B_ * T_ * OUT_;        // [B][T][512]
  float* renc = hid + (size_t)B_ * T_ * H_;          // [B][64]

  char* ws = (char*)d_ws;
  __bf16* Wc   = (__bf16*)(ws);                      // 1536*672*2      = 2,064,384 B
  __bf16* OutW = (__bf16*)(ws + 2064384);            // 48*512*2        = 49,152 B
  __bf16* Xbf  = (__bf16*)(ws + 2113536);            // 1024*120*160*2  = 39,321,600 B
  __bf16* Hbf0 = (__bf16*)(ws + 41435136);           // 1024*512*2 x2 ping-pong
  __bf16* Hbf1 = Hbf0 + (size_t)B_ * H_;

  rule_enc_kernel<<<B_ / 8, 256, 0, stream>>>(task_rule, rule_tf, ew1, eb1, ew2, eb2, renc);
  prep_weights<<<1024, 256, 0, stream>>>(w_ih, w_hh, out_w, Wc, OutW);
  prep_x<<<2048, 256, 0, stream>>>(x, renc, Xbf);
  init_h<<<1024, 256, 0, stream>>>(Hbf0);

  for (int t = 0; t < T_; ++t) {
    const __bf16* hp = (t & 1) ? Hbf1 : Hbf0;
    __bf16*       hn = (t & 1) ? Hbf0 : Hbf1;
    gru_step<<<dim3(B_ / 64, H_ / 64), 256, 0, stream>>>(Wc, Xbf, hp, hn, b_ih, b_hh, hid, t);
  }

  out_proj<<<(B_ * T_) / 128, 256, 0, stream>>>(hid, OutW, out_b, out);
}
